// _DecoderBlock_21964462752255
// MI455X (gfx1250) — matrix operand order repaired and checked by reading, not compiled, not run on hardware
//
#include <hip/hip_runtime.h>
#include <hip/hip_bf16.h>
#include <math.h>

typedef __bf16 bf16_t;
typedef __bf16 v16bf __attribute__((ext_vector_type(16)));
typedef __bf16 v8bf  __attribute__((ext_vector_type(8)));
typedef float  v8f   __attribute__((ext_vector_type(8)));

#define TILE_T        16
#define TILES_PER_WG  8
#define NTHREADS      256

// ---- LDS layout (bytes) ----
#define OFF_W1   0        // bf16 [4][64][128] = 65536 B
#define OFF_W2   65536    // bf16 [4][64][64]  = 32768 B
#define OFF_BUF  98304    // 32768 B time-shared: comb bf16 [8][16][128] -> out1 f32 [16][64][8] -> res f32 [16][64][8]
#define OFF_MID  131072   // bf16 [8][16][64]  = 16384 B
#define OFF_ROT  147456   // f32  [64][4]      = 1024 B
#define LDS_BYTES 148480

__global__ __launch_bounds__(NTHREADS, 2)
void clifford_block_kernel(const float* __restrict__ x,
                           const float* __restrict__ skip,
                           const float* __restrict__ W1,
                           const float* __restrict__ b1,
                           const float* __restrict__ gamma,
                           const float* __restrict__ biv,
                           const float* __restrict__ W2,
                           const float* __restrict__ b2,
                           float* __restrict__ out)
{
    __shared__ __align__(16) unsigned char smem[LDS_BYTES];
    bf16_t* sW1   = (bf16_t*)(smem + OFF_W1);
    bf16_t* sW2   = (bf16_t*)(smem + OFF_W2);
    bf16_t* sComb = (bf16_t*)(smem + OFF_BUF);   // [d][r][i]  i contiguous (128)
    float*  sOut1 = (float*) (smem + OFF_BUF);   // [r][o][d]
    float*  sRes  = (float*) (smem + OFF_BUF);   // [r][o][d]
    bf16_t* sMid  = (bf16_t*)(smem + OFF_MID);   // [d][r][c]  c contiguous (64)
    float*  sRot  = (float*) (smem + OFF_ROT);   // [c][4] = {cos, sinc*b0, sinc*b1, sinc*b2}

    const int tid  = threadIdx.x;
    const int lane = tid & 31;
    const int d    = tid >> 5;          // this wave's blade (0..7)
    const int g    = (d + 2) / 3;       // grade: 0,1,1,1,2,2,2,3
    const int mrow = lane & 15;
    const int half = lane >> 4;

    // ---------------- Phase A: weights + rotor coeffs -> LDS (once) ----------------
    for (int n = tid; n < 4*64*128; n += NTHREADS) sW1[n] = (bf16_t)W1[n];
    for (int n = tid; n < 4*64*64;  n += NTHREADS) sW2[n] = (bf16_t)W2[n];
    if (tid < 64) {
        float bx = biv[tid*3+0], by = biv[tid*3+1], bz = biv[tid*3+2];
        float th = sqrtf(bx*bx + by*by + bz*bz);
        float snc = (th > 1e-12f) ? (sinf(th) / th) : 1.0f;  // jnp.sinc(th/pi)
        sRot[tid*4+0] = cosf(th);
        sRot[tid*4+1] = snc * bx;
        sRot[tid*4+2] = snc * by;
        sRot[tid*4+3] = snc * bz;
    }
    __syncthreads();

    for (int it = 0; it < TILES_PER_WG; ++it) {
        const int t0 = (blockIdx.x * TILES_PER_WG + it) * TILE_T;
        const float* xt = x    + (size_t)t0 * 512;
        const float* st = skip + (size_t)t0 * 512;

        // -------- Phase B: stage activations (f32 -> bf16, blade-major) --------
        #pragma unroll 4
        for (int e = tid; e < TILE_T*512; e += NTHREADS) {
            int r = e >> 9, rem = e & 511, i = rem >> 3, dd = rem & 7;
            sComb[dd*2048 + r*128 + i]      = (bf16_t)xt[e];     // channels 0..63
            sComb[dd*2048 + r*128 + 64 + i] = (bf16_t)st[e];     // channels 64..127
        }
        if (it + 1 < TILES_PER_WG) {
            __builtin_prefetch(xt + TILE_T*512 + tid*32, 0, 0);  // next tile
            __builtin_prefetch(st + TILE_T*512 + tid*32, 0, 0);
        }
        __syncthreads();

        // -------- Phase C: stage-1 GEMM  out1[t,o,d] = comb[t,:,d] @ W1[g(d)].T --------
        v8f acc[4];
        #pragma unroll
        for (int n = 0; n < 4; ++n) acc[n] = (v8f){0,0,0,0,0,0,0,0};
        {
            const bf16_t* arow = sComb + d*2048 + mrow*128;
            #pragma unroll
            for (int k0 = 0; k0 < 128; k0 += 32) {
                v8bf alo = *(const v8bf*)(arow + k0 + half*8);
                v8bf ahi = *(const v8bf*)(arow + k0 + 16 + half*8);
                v16bf afrag = __builtin_shufflevector(alo, ahi, 0,1,2,3,4,5,6,7,8,9,10,11,12,13,14,15);
                #pragma unroll
                for (int n = 0; n < 4; ++n) {
                    const bf16_t* brow = sW1 + g*8192 + (n*16 + mrow)*128 + k0 + half*8;
                    v8bf blo = *(const v8bf*)(brow);
                    v8bf bhi = *(const v8bf*)(brow + 16);
                    v16bf bfrag = __builtin_shufflevector(blo, bhi, 0,1,2,3,4,5,6,7,8,9,10,11,12,13,14,15);
                    acc[n] = __builtin_amdgcn_wmma_f32_16x16x32_bf16(
                        false, afrag, false, bfrag, (short)0, acc[n], false, false);
                }
            }
        }
        __syncthreads();   // comb region reads done -> safe to overwrite as out1
        #pragma unroll
        for (int n = 0; n < 4; ++n) {
            int o = n*16 + mrow;
            float bv = b1[o*8 + d];
            #pragma unroll
            for (int vr = 0; vr < 8; ++vr) {
                int M = vr + half*8;
                sOut1[(M*64 + o)*8 + d] = acc[n][vr] + bv;
            }
        }
        __syncthreads();

        // -------- Phase D: grade-LayerNorm -> geometric GELU -> rotor sandwich --------
        #pragma unroll
        for (int pp = 0; pp < 4; ++pp) {
            int p = tid + pp*NTHREADS;          // (r,c) pair, 1024 total
            int r = p >> 6, c = p & 63;
            const float* yp = sOut1 + (r*64 + c)*8;
            float y0=yp[0],y1=yp[1],y2=yp[2],y3=yp[3],y4=yp[4],y5=yp[5],y6=yp[6],y7=yp[7];
            // grade-wise RMS
            float i0 = rsqrtf(y0*y0 + 1e-6f);
            float i1 = rsqrtf((y1*y1 + y2*y2 + y3*y3) * (1.0f/3.0f) + 1e-6f);
            float i2 = rsqrtf((y4*y4 + y5*y5 + y6*y6) * (1.0f/3.0f) + 1e-6f);
            float i3 = rsqrtf(y7*y7 + 1e-6f);
            float gm0 = gamma[c*4+0], gm1 = gamma[c*4+1], gm2 = gamma[c*4+2], gm3 = gamma[c*4+3];
            float z0 = y0*i0*gm0;
            float z1 = y1*i1*gm1, z2 = y2*i1*gm1, z3 = y3*i1*gm1;
            float z4 = y4*i2*gm2, z5 = y5*i2*gm2, z6 = y6*i2*gm2;
            float z7 = y7*i3*gm3;
            // GELU gate from scalar blade
            float gate = 0.5f * (1.0f + erff(z0 * 0.70710678118654752f));
            z0*=gate; z1*=gate; z2*=gate; z3*=gate; z4*=gate; z5*=gate; z6*=gate; z7*=gate;
            // rotor sandwich R z R~  (R = (R0; e12:R4, e13:R5, e23:R6))
            float R0 = sRot[c*4+0], R4 = sRot[c*4+1], R5 = sRot[c*4+2], R6 = sRot[c*4+3];
            float t0 = R0*z0 - R4*z4 - R5*z5 - R6*z6;
            float t1 = R0*z1 + R4*z2 + R5*z3 - R6*z7;
            float t2 = R0*z2 - R4*z1 + R5*z7 + R6*z3;
            float t3 = R0*z3 - R4*z7 - R5*z1 - R6*z2;
            float t4 = R0*z4 + R4*z0 - R5*z6 + R6*z5;
            float t5 = R0*z5 + R4*z6 + R5*z0 - R6*z4;
            float t6 = R0*z6 - R4*z5 + R5*z4 + R6*z0;
            float t7 = R0*z7 + R4*z3 - R5*z2 + R6*z1;
            float u0 = t0*R0 + t4*R4 + t5*R5 + t6*R6;
            float u1 = t1*R0 + t2*R4 + t3*R5 - t7*R6;
            float u2 = t2*R0 - t1*R4 - t7*R5 + t3*R6;
            float u3 = t3*R0 + t7*R4 - t1*R5 - t2*R6;
            float u4 = t4*R0 - t0*R4 - t6*R5 + t5*R6;
            float u5 = t5*R0 + t6*R4 - t0*R5 - t4*R6;
            float u6 = t6*R0 - t5*R4 + t4*R5 - t0*R6;
            float u7 = t7*R0 - t3*R4 + t2*R5 - t1*R6;
            bf16_t* mp = sMid + r*64 + c;               // [d][r][c], d-stride 1024
            mp[0*1024]=(bf16_t)u0; mp[1*1024]=(bf16_t)u1; mp[2*1024]=(bf16_t)u2; mp[3*1024]=(bf16_t)u3;
            mp[4*1024]=(bf16_t)u4; mp[5*1024]=(bf16_t)u5; mp[6*1024]=(bf16_t)u6; mp[7*1024]=(bf16_t)u7;
        }
        __syncthreads();

        // -------- Phase E: stage-2 GEMM  out[t,o,d] = mid[t,:,d] @ W2[g(d)].T --------
        v8f acc2[4];
        #pragma unroll
        for (int n = 0; n < 4; ++n) acc2[n] = (v8f){0,0,0,0,0,0,0,0};
        {
            const bf16_t* arow = sMid + d*1024 + mrow*64;
            #pragma unroll
            for (int k0 = 0; k0 < 64; k0 += 32) {
                v8bf alo = *(const v8bf*)(arow + k0 + half*8);
                v8bf ahi = *(const v8bf*)(arow + k0 + 16 + half*8);
                v16bf afrag = __builtin_shufflevector(alo, ahi, 0,1,2,3,4,5,6,7,8,9,10,11,12,13,14,15);
                #pragma unroll
                for (int n = 0; n < 4; ++n) {
                    const bf16_t* brow = sW2 + g*4096 + (n*16 + mrow)*64 + k0 + half*8;
                    v8bf blo = *(const v8bf*)(brow);
                    v8bf bhi = *(const v8bf*)(brow + 16);
                    v16bf bfrag = __builtin_shufflevector(blo, bhi, 0,1,2,3,4,5,6,7,8,9,10,11,12,13,14,15);
                    acc2[n] = __builtin_amdgcn_wmma_f32_16x16x32_bf16(
                        false, afrag, false, bfrag, (short)0, acc2[n], false, false);
                }
            }
        }
        #pragma unroll
        for (int n = 0; n < 4; ++n) {
            int o = n*16 + mrow;
            float bv = b2[o*8 + d];
            #pragma unroll
            for (int vr = 0; vr < 8; ++vr) {
                int M = vr + half*8;
                sRes[(M*64 + o)*8 + d] = acc2[n][vr] + bv;
            }
        }
        __syncthreads();

        // -------- Phase F: coalesced store --------
        {
            const float4* s4 = (const float4*)sRes;
            float4* o4 = (float4*)(out + (size_t)t0 * 512);
            #pragma unroll 2
            for (int e = tid; e < TILE_T*512/4; e += NTHREADS) o4[e] = s4[e];
        }
        __syncthreads();   // res region consumed before next tile's Phase B
    }
}

extern "C" void kernel_launch(void* const* d_in, const int* in_sizes, int n_in,
                              void* d_out, int out_size, void* d_ws, size_t ws_size,
                              hipStream_t stream) {
    const float* x     = (const float*)d_in[0];
    const float* skip  = (const float*)d_in[1];
    const float* W1    = (const float*)d_in[2];
    const float* b1    = (const float*)d_in[3];
    const float* gamma = (const float*)d_in[4];
    const float* biv   = (const float*)d_in[5];
    const float* W2    = (const float*)d_in[6];
    const float* b2    = (const float*)d_in[7];
    float* out = (float*)d_out;

    const int T = in_sizes[0] / 512;                 // B*N = 65536
    const int grid = T / (TILE_T * TILES_PER_WG);    // 512 workgroups
    clifford_block_kernel<<<grid, NTHREADS, 0, stream>>>(x, skip, W1, b1, gamma, biv, W2, b2, out);
}
